// EncoderLayer_81226421502316
// MI455X (gfx1250) — matrix operand order repaired and checked by reading, not compiled, not run on hardware
//
#include <hip/hip_runtime.h>
#include <hip/hip_bf16.h>

typedef _Float16 half_t;
typedef __attribute__((ext_vector_type(16))) _Float16 v16h;
typedef __attribute__((ext_vector_type(8)))  _Float16 v8h;
typedef __attribute__((ext_vector_type(8)))  float    v8f;

#define D_MODEL 512
#define D_FF    2048
#define SEQ_N   384
#define BATCH_B 2
#define ROWS    768   // B*N
#define EPS_LN  1e-6f

// ---------------------------------------------------------------------------
// fp32 -> fp16 conversion (weights, one pass at launch)
// ---------------------------------------------------------------------------
__global__ void f32_to_f16_kernel(const float* __restrict__ in,
                                  half_t* __restrict__ out, int n) {
    int i = blockIdx.x * 256 + threadIdx.x;
    if (i < n) out[i] = (half_t)in[i];
}

// ---------------------------------------------------------------------------
// Embedding gather * sqrt(D) + sinusoidal positional encoding.
// ---------------------------------------------------------------------------
__global__ void embed_pos_kernel(const int* __restrict__ tokens,
                                 const float* __restrict__ emb,
                                 float* __restrict__ xf,
                                 half_t* __restrict__ xh) {
    const int row = blockIdx.x;            // 0..ROWS-1
    const int n   = row % SEQ_N;           // position
    const int tok = tokens[row];
    const float pos = (float)n;
    const float sqrtd = 22.62741699796952f;       // sqrt(512)
    const float ln1e4 = 9.210340371976184f;       // ln(10000)
    for (int d = threadIdx.x; d < D_MODEL; d += 128) {
        float e = emb[(long)tok * D_MODEL + d] * sqrtd;
        int dpair = d & ~1;
        float inv = __expf(-((float)dpair / (float)D_MODEL) * ln1e4);
        float ang = pos * inv;
        float pe  = (d & 1) ? __cosf(ang) : __sinf(ang);
        float v = e + pe;
        xf[(long)row * D_MODEL + d] = v;
        xh[(long)row * D_MODEL + d] = (half_t)v;
    }
}

// ---------------------------------------------------------------------------
// Tiled WMMA GEMM: C[M,Ncols] = act(A[M,K] @ B[K,Ncols] + bias)
//   f16 operands, fp32 accumulation via v_wmma_f32_16x16x32_f16.
//   Block tile 128(M) x 64(N), BLK_K = 64 -> 8 WMMAs per barrier pair.
//   A tile staged with GLOBAL_LOAD_ASYNC_TO_LDS_B128 (ASYNCcnt path);
//   B tile staged via registers with a paired-k transpose so LDS stores are
//   dword-wide; each lane's fragment is two contiguous 16B LDS reads
//   in contiguous k order (k = 16*half + s), same for A and B.
// ---------------------------------------------------------------------------
#define BLK_M 128
#define BLK_N 64
#define BLK_K 64
#define LDS_STRIDE 72   // 64 + 8 halves pad: 16B-aligned rows, conflict-free
#define SMEM_A_BYTES (BLK_M * LDS_STRIDE * 2)
#define SMEM_B_BYTES (BLK_N * LDS_STRIDE * 2)
#define SMEM_BYTES   (SMEM_A_BYTES + SMEM_B_BYTES)

__global__ __launch_bounds__(256)
void gemm_wmma_f16(const half_t* __restrict__ A, const half_t* __restrict__ B,
                   const float* __restrict__ bias,
                   float* __restrict__ Cf, half_t* __restrict__ Ch,
                   int M, int Ncols, int K,
                   long sA, long sB, long sC, int relu) {
    extern __shared__ __align__(16) char smem[];
    half_t* As = (half_t*)smem;                    // [128][LDS_STRIDE]
    half_t* Bt = (half_t*)(smem + SMEM_A_BYTES);   // [64][LDS_STRIDE] transposed

    const int t    = threadIdx.x;
    const int wave = t >> 5;
    const int lane = t & 31;
    const int wm   = wave & 3;       // 4 wave-rows of 32
    const int wn   = wave >> 2;      // 2 wave-cols of 32
    const int sel  = lane >> 4;
    const int l15  = lane & 15;

    const int bm = blockIdx.y * BLK_M;
    const int bn = blockIdx.x * BLK_N;
    const int batch = blockIdx.z;
    A += (long)batch * sA;
    B += (long)batch * sB;
    const unsigned long long Abase = (unsigned long long)A;

    v8f acc[2][2] = {};

    for (int kb = 0; kb < K; kb += BLK_K) {
        // --- A tile (128 x 64 halves = 16KB): async copy global -> LDS.
        //     1024 x 16B chunks, 4 per thread; ASYNCcnt-tracked.
        #pragma unroll
        for (int i = 0; i < 4; ++i) {
            int c   = t + i * 256;            // 0..1023
            int row = c >> 3;                 // 0..127
            int kc  = (c & 7) * 8;            // 0..56 halves
            unsigned lds_off = (unsigned)(unsigned long long)(&As[row * LDS_STRIDE + kc]);
            int      goff    = (int)((((long)(bm + row)) * K + kb + kc) * 2);
            asm volatile("global_load_async_to_lds_b128 %0, %1, %2"
                         :: "v"(lds_off), "v"(goff), "s"(Abase)
                         : "memory");
        }
        // --- B tile (64 x 64 halves = 8KB): each thread loads 2 adjacent
        //     k-rows x 8 cols, packs k-pairs, stores 8 dwords to Bt[col][k].
        {
            int brow = (t >> 3) * 2;          // even K index 0..62
            int bcol = (t & 7) * 8;           // col start 0..56
            const half_t* src = B + (long)(kb + brow) * Ncols + bn + bcol;
            if (kb + BLK_K < K)               // prefetch next K-stage of B
                __builtin_prefetch(src + (long)BLK_K * Ncols, 0, 1);
            half_t t0[8], t1[8];
            *(uint4*)t0 = *(const uint4*)src;
            *(uint4*)t1 = *(const uint4*)(src + Ncols);
            #pragma unroll
            for (int j = 0; j < 8; ++j) {
                half_t pair[2] = { t0[j], t1[j] };
                *(unsigned*)(&Bt[(bcol + j) * LDS_STRIDE + brow]) = *(unsigned*)pair;
            }
        }
        asm volatile("s_wait_asynccnt 0x0" ::: "memory");
        __syncthreads();

        // --- two 32-deep WMMA stages out of the 64-deep tile
        #pragma unroll
        for (int kk = 0; kk < BLK_K; kk += 32) {
            v16h af[2], bf[2];
            #pragma unroll
            for (int mi = 0; mi < 2; ++mi) {
                int mrow  = wm * 32 + mi * 16 + l15;
                int base0 = kk + sel * 16;
                int base1 = kk + sel * 16 + 8;
                v8h lo = *(const v8h*)(&As[mrow * LDS_STRIDE + base0]);
                v8h hi = *(const v8h*)(&As[mrow * LDS_STRIDE + base1]);
                #pragma unroll
                for (int e = 0; e < 8; ++e) { af[mi][e] = lo[e]; af[mi][8 + e] = hi[e]; }
            }
            #pragma unroll
            for (int ni = 0; ni < 2; ++ni) {
                int col   = wn * 32 + ni * 16 + l15;
                int kbase = kk + sel * 16;
                v8h lo = *(const v8h*)(&Bt[col * LDS_STRIDE + kbase]);
                v8h hi = *(const v8h*)(&Bt[col * LDS_STRIDE + kbase + 8]);
                #pragma unroll
                for (int e = 0; e < 8; ++e) { bf[ni][e] = lo[e]; bf[ni][8 + e] = hi[e]; }
            }
            #pragma unroll
            for (int mi = 0; mi < 2; ++mi)
                #pragma unroll
                for (int ni = 0; ni < 2; ++ni)
                    acc[mi][ni] = __builtin_amdgcn_wmma_f32_16x16x32_f16(
                        false, af[mi], false, bf[ni],
                        (short)0, acc[mi][ni], false, false);
        }
        __syncthreads();
    }

    // --- epilogue: bias (+ReLU), write fp32 and/or f16 outputs
    float*  cf = Cf ? Cf + (long)batch * sC : nullptr;
    half_t* ch = Ch ? Ch + (long)batch * sC : nullptr;
    #pragma unroll
    for (int ni = 0; ni < 2; ++ni) {
        int col = bn + wn * 32 + ni * 16 + l15;
        float bv = bias ? bias[col] : 0.0f;
        #pragma unroll
        for (int mi = 0; mi < 2; ++mi) {
            #pragma unroll
            for (int r = 0; r < 8; ++r) {
                int row = bm + wm * 32 + mi * 16 + (sel ? 8 : 0) + r;
                float v = acc[mi][ni][r] + bv;
                if (relu) v = fmaxf(v, 0.0f);
                long idx = (long)row * Ncols + col;
                if (cf) cf[idx] = v;
                if (ch) ch[idx] = (half_t)v;
            }
        }
    }
}

// ---------------------------------------------------------------------------
// Additive-attention scores + row softmax.
//   scores[i,j] = sum_d scale[d] * tanh(q[i,d] + k[j,d])
//   tanh via 1 - 2/(exp(2x)+1): one v_exp_f32 (TRANS pipe) + fast div,
//   exact in both saturation limits. float4 VMEM reads (rows 2KB aligned).
// ---------------------------------------------------------------------------
__device__ __forceinline__ float fast_tanh(float x) {
    float e = __expf(2.0f * x);
    return 1.0f - __fdividef(2.0f, e + 1.0f);
}

__global__ __launch_bounds__(128)
void scores_softmax_kernel(const float* __restrict__ q,
                           const float* __restrict__ k,
                           const float* __restrict__ scale,
                           half_t* __restrict__ wout) {
    const int i = blockIdx.x;          // global row 0..ROWS-1
    const int b = i / SEQ_N;
    const int t = threadIdx.x;

    __shared__ __align__(16) float qs[D_MODEL];
    __shared__ __align__(16) float sc[D_MODEL];
    __shared__ float red[128];

    for (int d = t; d < D_MODEL; d += 128) {
        qs[d] = q[(long)i * D_MODEL + d];
        sc[d] = scale[d];
    }
    __syncthreads();

    float a[3];
    #pragma unroll
    for (int jj = 0; jj < 3; ++jj) {
        int j = jj * 128 + t;
        const float4* kr4 = (const float4*)(k + ((long)(b * SEQ_N + j)) * D_MODEL);
        float acc = 0.0f;
        #pragma unroll 2
        for (int d4 = 0; d4 < D_MODEL / 4; ++d4) {
            float4 kv = kr4[d4];
            float4 qv = *(const float4*)&qs[d4 * 4];
            float4 sv = *(const float4*)&sc[d4 * 4];
            acc += sv.x * fast_tanh(qv.x + kv.x);
            acc += sv.y * fast_tanh(qv.y + kv.y);
            acc += sv.z * fast_tanh(qv.z + kv.z);
            acc += sv.w * fast_tanh(qv.w + kv.w);
        }
        a[jj] = acc;
    }

    // block max
    float mloc = fmaxf(a[0], fmaxf(a[1], a[2]));
    red[t] = mloc; __syncthreads();
    for (int off = 64; off > 0; off >>= 1) {
        if (t < off) red[t] = fmaxf(red[t], red[t + off]);
        __syncthreads();
    }
    float m = red[0]; __syncthreads();

    float e0 = __expf(a[0] - m), e1 = __expf(a[1] - m), e2 = __expf(a[2] - m);
    red[t] = e0 + e1 + e2; __syncthreads();
    for (int off = 64; off > 0; off >>= 1) {
        if (t < off) red[t] += red[t + off];
        __syncthreads();
    }
    float inv = 1.0f / red[0];

    long base = (long)i * SEQ_N;
    wout[base + t]       = (half_t)(e0 * inv);
    wout[base + t + 128] = (half_t)(e1 * inv);
    wout[base + t + 256] = (half_t)(e2 * inv);
}

// ---------------------------------------------------------------------------
// Residual add + LayerNorm. One 128-thread block per row of 512.
// ---------------------------------------------------------------------------
__global__ __launch_bounds__(128)
void add_layernorm_kernel(const float* __restrict__ xa,
                          const float* __restrict__ xb,
                          const float* __restrict__ gamma,
                          const float* __restrict__ beta,
                          float* __restrict__ outf,
                          half_t* __restrict__ outh) {
    const int row = blockIdx.x;
    const int t = threadIdx.x;
    __shared__ float rsum[128];
    __shared__ float rsq[128];

    float v[4];
    float s = 0.0f, ss = 0.0f;
    #pragma unroll
    for (int c = 0; c < 4; ++c) {
        int d = c * 128 + t;
        float x = xa[(long)row * D_MODEL + d] + xb[(long)row * D_MODEL + d];
        v[c] = x; s += x; ss += x * x;
    }
    rsum[t] = s; rsq[t] = ss; __syncthreads();
    for (int off = 64; off > 0; off >>= 1) {
        if (t < off) { rsum[t] += rsum[t + off]; rsq[t] += rsq[t + off]; }
        __syncthreads();
    }
    float mu  = rsum[0] * (1.0f / D_MODEL);
    float var = rsq[0] * (1.0f / D_MODEL) - mu * mu;
    float rstd = rsqrtf(var + EPS_LN);
    #pragma unroll
    for (int c = 0; c < 4; ++c) {
        int d = c * 128 + t;
        float y = (v[c] - mu) * rstd * gamma[d] + beta[d];
        outf[(long)row * D_MODEL + d] = y;
        if (outh) outh[(long)row * D_MODEL + d] = (half_t)y;
    }
}

// ---------------------------------------------------------------------------
extern "C" void kernel_launch(void* const* d_in, const int* in_sizes, int n_in,
                              void* d_out, int out_size, void* d_ws, size_t ws_size,
                              hipStream_t stream) {
    (void)in_sizes; (void)n_in; (void)out_size; (void)ws_size;

    const int*   tokens = (const int*)  d_in[0];
    const float* emb    = (const float*)d_in[1];
    const float* Wq     = (const float*)d_in[2];
    const float* bq     = (const float*)d_in[3];
    const float* Wk     = (const float*)d_in[4];
    const float* bk     = (const float*)d_in[5];
    const float* Wv     = (const float*)d_in[6];
    const float* bv     = (const float*)d_in[7];
    const float* ascale = (const float*)d_in[8];
    const float* Wo     = (const float*)d_in[9];
    const float* bo     = (const float*)d_in[10];
    const float* g1     = (const float*)d_in[11];
    const float* beta1  = (const float*)d_in[12];
    const float* W1     = (const float*)d_in[13];
    const float* bf1    = (const float*)d_in[14];
    const float* W2     = (const float*)d_in[15];
    const float* bf2    = (const float*)d_in[16];
    const float* g2     = (const float*)d_in[17];
    const float* beta2  = (const float*)d_in[18];

    // workspace bump allocator (256B aligned)
    char* base = (char*)d_ws;
    size_t off = 0;
    auto alloc = [&](size_t bytes) -> void* {
        void* p = base + off;
        off = (off + bytes + 255) & ~(size_t)255;
        return p;
    };

    float*  xf    = (float*) alloc((size_t)ROWS * D_MODEL * 4);
    half_t* xh    = (half_t*)alloc((size_t)ROWS * D_MODEL * 2);
    float*  qf    = (float*) alloc((size_t)ROWS * D_MODEL * 4);
    float*  kf    = (float*) alloc((size_t)ROWS * D_MODEL * 4);
    half_t* vh    = (half_t*)alloc((size_t)ROWS * D_MODEL * 2);
    half_t* wq_h  = (half_t*)alloc((size_t)D_MODEL * D_MODEL * 2);
    half_t* wk_h  = (half_t*)alloc((size_t)D_MODEL * D_MODEL * 2);
    half_t* wv_h  = (half_t*)alloc((size_t)D_MODEL * D_MODEL * 2);
    half_t* wo_h  = (half_t*)alloc((size_t)D_MODEL * D_MODEL * 2);
    half_t* w1_h  = (half_t*)alloc((size_t)D_MODEL * D_FF * 2);
    half_t* w2_h  = (half_t*)alloc((size_t)D_FF * D_MODEL * 2);
    half_t* wts_h = (half_t*)alloc((size_t)BATCH_B * SEQ_N * SEQ_N * 2);
    half_t* attn_h= (half_t*)alloc((size_t)ROWS * D_MODEL * 2);
    float*  projf = (float*) alloc((size_t)ROWS * D_MODEL * 4);
    float*  x1f   = (float*) alloc((size_t)ROWS * D_MODEL * 4);
    half_t* x1h   = (half_t*)alloc((size_t)ROWS * D_MODEL * 2);
    half_t* hh    = (half_t*)alloc((size_t)ROWS * D_FF * 2);
    float*  yf    = (float*) alloc((size_t)ROWS * D_MODEL * 4);

    // 1) weight fp32 -> f16
    {
        int n = D_MODEL * D_MODEL;
        int g = (n + 255) / 256;
        f32_to_f16_kernel<<<g, 256, 0, stream>>>(Wq, wq_h, n);
        f32_to_f16_kernel<<<g, 256, 0, stream>>>(Wk, wk_h, n);
        f32_to_f16_kernel<<<g, 256, 0, stream>>>(Wv, wv_h, n);
        f32_to_f16_kernel<<<g, 256, 0, stream>>>(Wo, wo_h, n);
        int nf = D_MODEL * D_FF;
        int gf = (nf + 255) / 256;
        f32_to_f16_kernel<<<gf, 256, 0, stream>>>(W1, w1_h, nf);
        f32_to_f16_kernel<<<gf, 256, 0, stream>>>(W2, w2_h, nf);
    }

    // 2) embedding + positional encoding
    embed_pos_kernel<<<ROWS, 128, 0, stream>>>(tokens, emb, xf, xh);

    // 3) Q, K (fp32 outputs for tanh path), V (f16 output for attn@V)
    dim3 gQKV(D_MODEL / BLK_N, ROWS / BLK_M, 1);
    gemm_wmma_f16<<<gQKV, 256, SMEM_BYTES, stream>>>(xh, wq_h, bq, qf, nullptr,
                                            ROWS, D_MODEL, D_MODEL, 0, 0, 0, 0);
    gemm_wmma_f16<<<gQKV, 256, SMEM_BYTES, stream>>>(xh, wk_h, bk, kf, nullptr,
                                            ROWS, D_MODEL, D_MODEL, 0, 0, 0, 0);
    gemm_wmma_f16<<<gQKV, 256, SMEM_BYTES, stream>>>(xh, wv_h, bv, nullptr, vh,
                                            ROWS, D_MODEL, D_MODEL, 0, 0, 0, 0);

    // 4) additive-attention scores + softmax -> f16 weights
    scores_softmax_kernel<<<ROWS, 128, 0, stream>>>(qf, kf, ascale, wts_h);

    // 5) attn = weights @ V   (batched over B)
    dim3 gAV(D_MODEL / BLK_N, SEQ_N / BLK_M, BATCH_B);
    gemm_wmma_f16<<<gAV, 256, SMEM_BYTES, stream>>>(wts_h, vh, nullptr, nullptr, attn_h,
                                           SEQ_N, D_MODEL, SEQ_N,
                                           (long)SEQ_N * SEQ_N,
                                           (long)SEQ_N * D_MODEL,
                                           (long)SEQ_N * D_MODEL, 0);

    // 6) output projection
    gemm_wmma_f16<<<gQKV, 256, SMEM_BYTES, stream>>>(attn_h, wo_h, bo, projf, nullptr,
                                            ROWS, D_MODEL, D_MODEL, 0, 0, 0, 0);

    // 7) residual + LN1
    add_layernorm_kernel<<<ROWS, 128, 0, stream>>>(xf, projf, g1, beta1, x1f, x1h);

    // 8) FFN1 (+ReLU, f16 out) and FFN2
    dim3 gF1(D_FF / BLK_N, ROWS / BLK_M, 1);
    gemm_wmma_f16<<<gF1, 256, SMEM_BYTES, stream>>>(x1h, w1_h, bf1, nullptr, hh,
                                           ROWS, D_FF, D_MODEL, 0, 0, 0, 1);
    gemm_wmma_f16<<<gQKV, 256, SMEM_BYTES, stream>>>(hh, w2_h, bf2, yf, nullptr,
                                            ROWS, D_MODEL, D_FF, 0, 0, 0, 0);

    // 9) residual + LN2 -> output (fp32)
    add_layernorm_kernel<<<ROWS, 128, 0, stream>>>(x1f, yf, g2, beta2,
                                                   (float*)d_out, nullptr);
}
